// SwinBlock_14989435863101
// MI455X (gfx1250) — matrix operand order repaired and checked by reading, not compiled, not run on hardware
//
#include <hip/hip_runtime.h>
#include <hip/hip_bf16.h>
#include <math.h>

typedef __attribute__((ext_vector_type(16))) __bf16 v16bf;
typedef __attribute__((ext_vector_type(16))) unsigned short v16us;
typedef __attribute__((ext_vector_type(8)))  unsigned short us8;
typedef __attribute__((ext_vector_type(8)))  float v8f;
typedef __attribute__((ext_vector_type(4)))  int i4x;

#define TOK     65536      // B_ * N = 1024 windows * 64 tokens
#define CDIM    512
#define INNER   1365
#define INNER_P 1376       // padded to multiple of 32 for WMMA K loop
#define FC1N    2730

#ifndef __has_builtin
#define __has_builtin(x) 0
#endif
#if __has_builtin(__builtin_amdgcn_global_load_async_to_lds_b128)
#define USE_ASYNC_LDS 1
#else
#define USE_ASYNC_LDS 0
#endif

#if USE_ASYNC_LDS
// builtin expects (int4 AS1*, int4 AS3*, imm offset, imm cpol) per hipcc diagnostic
#define ASYNC_G2L_B128(gp, lp)                                            \
  __builtin_amdgcn_global_load_async_to_lds_b128(                         \
      (__attribute__((address_space(1))) i4x*)(gp),                       \
      (__attribute__((address_space(3))) i4x*)(lp), 0, 0)
#endif

static __device__ __forceinline__ void wait_async0() {
#if __has_builtin(__builtin_amdgcn_s_wait_asynccnt)
  __builtin_amdgcn_s_wait_asynccnt(0);
#else
  asm volatile("s_wait_asynccnt 0" ::: "memory");
#endif
}

static __device__ __forceinline__ unsigned short f2bf(float f) {
  union { float f; unsigned int u; } v; v.f = f;
  unsigned int x = v.u;
  x += 0x7FFFu + ((x >> 16) & 1u);          // round-to-nearest-even
  return (unsigned short)(x >> 16);
}
static __device__ __forceinline__ float bf2f(unsigned short h) {
  union { unsigned int u; float f; } v; v.u = ((unsigned int)h) << 16;
  return v.f;
}

// ---------------- adaLN modulation: mods[b, 0:3072] = silu(c) @ ada_w.T + ada_b
__global__ __launch_bounds__(64) void mods_kernel(const float* __restrict__ c,
                                                  const float* __restrict__ w,
                                                  const float* __restrict__ bb,
                                                  float* __restrict__ mods) {
  int g = blockIdx.x;                // 0..6143
  int b = g / 3072, j = g % 3072;
  int tid = threadIdx.x;
  float acc = 0.f;
  for (int k = tid; k < CDIM; k += 64) {
    float cv = c[b * CDIM + k];
    float s  = cv / (1.0f + expf(-cv));
    acc += s * w[(size_t)j * CDIM + k];
  }
  __shared__ float red[64];
  red[tid] = acc; __syncthreads();
  for (int st = 32; st > 0; st >>= 1) { if (tid < st) red[tid] += red[tid + st]; __syncthreads(); }
  if (tid == 0) mods[b * 3072 + j] = red[0] + bb[j];
}

// ---------------- QKV bias: concat(q_bias, 0, v_bias)
__global__ void biasq_kernel(const float* __restrict__ qb, const float* __restrict__ vb,
                             float* __restrict__ out) {
  int i = blockIdx.x * 256 + threadIdx.x;
  if (i >= 1536) return;
  float v = 0.f;
  if (i < 512) v = qb[i]; else if (i >= 1024) v = vb[i - 1024];
  out[i] = v;
}

// ---------------- f32 -> bf16 (optionally K-padded with zeros)
__global__ void cvt_pad_kernel(const float* __restrict__ src, unsigned short* __restrict__ dst,
                               int N, int Kin, int Kout) {
  long i = (long)blockIdx.x * 256 + threadIdx.x;
  if (i >= (long)N * Kout) return;
  int  k = (int)(i % Kout);
  long n = i / Kout;
  dst[i] = (k < Kin) ? f2bf(src[n * (long)Kin + k]) : (unsigned short)0;
}

// ---------------- CPB MLP: hbt[225,16] = relu(tbl@w1.T + b1) @ w2.T
__global__ __launch_bounds__(128) void cpb_kernel(const float* __restrict__ w1,
                                                  const float* __restrict__ b1,
                                                  const float* __restrict__ w2,
                                                  float* __restrict__ hbt) {
  int p = blockIdx.x, tid = threadIdx.x;
  int i = p / 15, j = p % 15;
  float v0 = (float)(i - 7) / 7.0f * 8.0f;
  float v1 = (float)(j - 7) / 7.0f * 8.0f;
  float t0 = copysignf(log2f(fabsf(v0) + 1.0f) / 3.0f, v0);   // /log2(8)
  float t1 = copysignf(log2f(fabsf(v1) + 1.0f) / 3.0f, v1);
  float acc[16];
#pragma unroll
  for (int o = 0; o < 16; ++o) acc[o] = 0.f;
  for (int u = tid; u < 512; u += 128) {
    float h = fmaxf(0.0f, t0 * w1[u * 2] + t1 * w1[u * 2 + 1] + b1[u]);
#pragma unroll
    for (int o = 0; o < 16; ++o) acc[o] += h * w2[o * 512 + u];
  }
  __shared__ float red[128][16];
#pragma unroll
  for (int o = 0; o < 16; ++o) red[tid][o] = acc[o];
  __syncthreads();
  for (int st = 64; st > 0; st >>= 1) {
    if (tid < st)
#pragma unroll
      for (int o = 0; o < 16; ++o) red[tid][o] += red[tid + st][o];
    __syncthreads();
  }
  if (tid < 16) hbt[p * 16 + tid] = red[0][tid];
}

// ---------------- rpb[h][n][m] = 16*sigmoid(hbt[idx(n,m), h])
__global__ void rpb_kernel(const float* __restrict__ hbt, float* __restrict__ rpb) {
  int i = blockIdx.x * 256 + threadIdx.x;   // 65536 = 16*64*64
  int h = i >> 12, n = (i >> 6) & 63, m = i & 63;
  int dh = (n >> 3) - (m >> 3) + 7;
  int dw = (n & 7) - (m & 7) + 7;
  float v = hbt[(dh * 15 + dw) * 16 + h];
  rpb[i] = 16.0f / (1.0f + expf(-v));
}

// ---------------- LN + adaLN modulate; windowed=1 also does roll(-4,-4)+partition
__global__ __launch_bounds__(128) void ln_mod_kernel(
    const float* __restrict__ xin, const float* __restrict__ g, const float* __restrict__ bvec,
    const float* __restrict__ mods, int scale_off, int shift_off, int windowed,
    unsigned short* __restrict__ outb) {
  int t = blockIdx.x;                 // destination row (window order if windowed)
  int b; long src;
  if (windowed) {
    int w = t >> 6, n = t & 63;
    b = w >> 9;
    int widx = w & 511;
    int wr = widx >> 5, wc = widx & 31;
    int oh = ((wr << 3) + (n >> 3) + 4) & 127;   // roll(-4): src row = (dst+4) mod H
    int ow = ((wc << 3) + (n & 7) + 4) & 255;
    src = (long)b * 32768 + oh * 256 + ow;
  } else { b = t >> 15; src = t; }
  const float* xr = xin + src * (long)CDIM;
  int tid = threadIdx.x;
  float vals[4], s = 0.f, sq = 0.f;
#pragma unroll
  for (int i = 0; i < 4; ++i) {
    float v = xr[tid + i * 128]; vals[i] = v; s += v; sq += v * v;
  }
  __shared__ float rs[128], rq[128];
  rs[tid] = s; rq[tid] = sq; __syncthreads();
  for (int st = 64; st > 0; st >>= 1) {
    if (tid < st) { rs[tid] += rs[tid + st]; rq[tid] += rq[tid + st]; }
    __syncthreads();
  }
  float mean = rs[0] * (1.0f / 512.0f);
  float var  = rq[0] * (1.0f / 512.0f) - mean * mean;
  float rstd = rsqrtf(var + 1e-5f);
  const float* mrow = mods + b * 3072;
#pragma unroll
  for (int i = 0; i < 4; ++i) {
    int cidx = tid + i * 128;
    float y = (vals[i] - mean) * rstd * g[cidx] + bvec[cidx];
    y = y * (1.0f + mrow[scale_off + cidx]) + mrow[shift_off + cidx];
    outb[(long)t * CDIM + cidx] = f2bf(y);
  }
}

// ---------------- cosine-sim window attention; one block per (window, head)
__global__ __launch_bounds__(64) void attn_kernel(
    const unsigned short* __restrict__ qkv, const float* __restrict__ rpb,
    const float* __restrict__ logit_scale, unsigned short* __restrict__ ctx) {
  int blk = blockIdx.x;
  int w = blk >> 4, h = blk & 15;
  int n = threadIdx.x;
  __shared__ float kn[64][33];
  __shared__ float vs[64][33];
  __shared__ float sc[64][65];
  const unsigned short* base = qkv + (long)(w * 64 + n) * 1536;
  float qr[32], kr[32], ssq = 0.f, ssk = 0.f;
#pragma unroll
  for (int d = 0; d < 32; ++d) {
    float qv = bf2f(base[h * 32 + d]);        qr[d] = qv; ssq += qv * qv;
    float kv = bf2f(base[512 + h * 32 + d]);  kr[d] = kv; ssk += kv * kv;
  }
  float iq = 1.0f / fmaxf(sqrtf(ssq), 1e-12f);
  float ik = 1.0f / fmaxf(sqrtf(ssk), 1e-12f);
#pragma unroll
  for (int d = 0; d < 32; ++d) {
    qr[d] *= iq;
    kn[n][d] = kr[d] * ik;
    vs[n][d] = bf2f(base[1024 + h * 32 + d]);
  }
  __syncthreads();
  float ls = expf(fminf(logit_scale[h], 4.60517018599f));   // log(100)
  bool edge = (((w & 511) >> 5) == 15);                     // mask_type 'h': only last window-row
  bool top  = (n >> 3) < 4;
  const float* rp = rpb + h * 4096 + n * 64;
  float mx = -1e30f;
  for (int m = 0; m < 64; ++m) {
    float dot = 0.f;
#pragma unroll
    for (int d = 0; d < 32; ++d) dot += qr[d] * kn[m][d];
    float val = dot * ls + rp[m];
    if (edge && (top != ((m >> 3) < 4))) val -= 100.0f;
    sc[n][m] = val; mx = fmaxf(mx, val);
  }
  float se = 0.f;
  for (int m = 0; m < 64; ++m) { float e = expf(sc[n][m] - mx); sc[n][m] = e; se += e; }
  float inv = 1.0f / se;
  float o[32];
#pragma unroll
  for (int d = 0; d < 32; ++d) o[d] = 0.f;
  for (int m = 0; m < 64; ++m) {
    float p = sc[n][m] * inv;
#pragma unroll
    for (int d = 0; d < 32; ++d) o[d] += p * vs[m][d];
  }
  unsigned short* cb = ctx + (long)(w * 64 + n) * CDIM + h * 32;
#pragma unroll
  for (int d = 0; d < 32; ++d) cb[d] = f2bf(o[d]);
}

// ---------------- GeGLU: hg[t,k] = gelu(fc1out[t,k+1365]) * fc1out[t,k], padded to 1376
__global__ void geglu_kernel(const unsigned short* __restrict__ fc1out,
                             unsigned short* __restrict__ hg) {
  long i = (long)blockIdx.x * 256 + threadIdx.x;
  if (i >= (long)TOK * INNER_P) return;
  int  k = (int)(i % INNER_P);
  long t = i / INNER_P;
  unsigned short r = 0;
  if (k < INNER) {
    float a  = bf2f(fc1out[t * FC1N + k]);
    float gt = bf2f(fc1out[t * FC1N + INNER + k]);
    float ge = 0.5f * gt * (1.0f + erff(gt * 0.70710678118f));   // exact gelu
    r = f2bf(ge * a);
  }
  hg[i] = r;
}

// ---------------- bf16 WMMA GEMM: C[M,N] = A[M,K] @ W[N,K]^T (f32 accum) + fused epilogue
// mode 0: +bias, store bf16 (QKV)           mode 2: store bf16, N-guard (FC1)
// mode 1: proj: window-reverse+roll, out = x + gm*(acc+bias)   mode 3: out += gp*acc (FC2)
#define BM 128
#define BN 128
#define BK 32
__global__ __launch_bounds__(256) void wmma_gemm_kernel(
    const unsigned short* __restrict__ A, int lda,
    const unsigned short* __restrict__ Bw, int ldb,
    int N, int K, int mode,
    const float* __restrict__ bias, const float* __restrict__ xres,
    const float* __restrict__ mods, float* __restrict__ outf,
    unsigned short* __restrict__ outb, int ldc) {
  __shared__ unsigned short As[2][BM][BK];   // 16 KB double-buffered
  __shared__ unsigned short Bs[2][BN][BK];   // 16 KB double-buffered
  const int tid  = threadIdx.x;
  const int lane = tid & 31;
  const int wave = tid >> 5;
  const int wm   = wave >> 1;     // 0..3 -> 32-row strip
  const int wn   = wave & 1;      // 0..1 -> 64-col strip
  const int bm   = blockIdx.y;
  const int bn   = blockIdx.x;
  const int rowBase = bm * BM;
  const int colBase = bn * BN;
  v8f acc[2][4] = {};

  // per-thread tile chunk coords: 512 us8 chunks per 128x32 tile, 2 per thread
  const int row0 = tid >> 2;                // chunk i=0
  const int row1 = (tid + 256) >> 2;        // chunk i=1
  const int c80  = (tid & 3) << 3;

  auto load_stage = [&](int s, int k0) {
#pragma unroll
    for (int i = 0; i < 2; ++i) {
      int row = i ? row1 : row0;
      // A tile
      const unsigned short* ga = A + (size_t)(rowBase + row) * lda + k0 + c80;
      unsigned short* la = &As[s][row][c80];
      // B tile (N-guarded)
      int ng = colBase + row;
      const unsigned short* gb = Bw + (size_t)ng * ldb + k0 + c80;
      unsigned short* lb = &Bs[s][row][c80];
#if USE_ASYNC_LDS
      ASYNC_G2L_B128(ga, la);
      if (ng < N) {
        ASYNC_G2L_B128(gb, lb);
      } else {
        us8 z{};
        *(us8*)lb = z;
      }
#else
      *(us8*)la = *(const us8*)ga;
      us8 vb{};
      if (ng < N) vb = *(const us8*)gb;
      *(us8*)lb = vb;
#endif
    }
  };

  const int nk = K / BK;
  load_stage(0, 0);
#if USE_ASYNC_LDS
  wait_async0();
#endif
  __syncthreads();

  for (int ki = 0; ki < nk; ++ki) {
    const int s = ki & 1;
    if (ki + 1 < nk) load_stage(s ^ 1, (ki + 1) * BK);

    const int r15 = lane & 15, kh = lane >> 4;
    v16us afr[2], bfr[4];
#pragma unroll
    for (int rt = 0; rt < 2; ++rt) {
      int row = wm * 32 + rt * 16 + r15;
      union { v16us v; us8 h[2]; } u;
      u.h[0] = *(const us8*)(&As[s][row][kh * 8]);       // VGPR0-3: K = kh*8 .. +7
      u.h[1] = *(const us8*)(&As[s][row][16 + kh * 8]);  // VGPR4-7: K = 16+kh*8 ..
      afr[rt] = u.v;
    }
#pragma unroll
    for (int ct = 0; ct < 4; ++ct) {
      int col = wn * 64 + ct * 16 + r15;
      union { v16us v; us8 h[2]; } u;
      u.h[0] = *(const us8*)(&Bs[s][col][kh * 8]);       // VGPR0-3: K = kh*8 .. +7
      u.h[1] = *(const us8*)(&Bs[s][col][16 + kh * 8]);  // VGPR4-7: K = 16+kh*8 ..
      bfr[ct] = u.v;
    }
#pragma unroll
    for (int rt = 0; rt < 2; ++rt)
#pragma unroll
      for (int ct = 0; ct < 4; ++ct)
        acc[rt][ct] = __builtin_amdgcn_wmma_f32_16x16x32_bf16(
            false, __builtin_bit_cast(v16bf, afr[rt]),
            false, __builtin_bit_cast(v16bf, bfr[ct]),
            (short)0, acc[rt][ct], false, false);

#if USE_ASYNC_LDS
    wait_async0();
#endif
    __syncthreads();
  }

  // epilogue: D layout = lane&15 -> N col; (lane>>4)*8 + vgpr -> M row
  const int r15 = lane & 15, half = lane >> 4;
#pragma unroll
  for (int rt = 0; rt < 2; ++rt)
#pragma unroll
    for (int ct = 0; ct < 4; ++ct)
#pragma unroll
      for (int j = 0; j < 8; ++j) {
        int rg = rowBase + wm * 32 + rt * 16 + half * 8 + j;
        int cg = colBase + wn * 64 + ct * 16 + r15;
        float v = acc[rt][ct][j];
        if (mode == 0) {
          outb[(size_t)rg * ldc + cg] = f2bf(v + bias[cg]);
        } else if (mode == 2) {
          if (cg < N) outb[(size_t)rg * ldc + cg] = f2bf(v);
        } else if (mode == 1) {
          int t = rg, w = t >> 6, n = t & 63;
          int b = w >> 9, widx = w & 511;
          int wr = widx >> 5, wc = widx & 31;
          int oh = ((wr << 3) + (n >> 3) + 4) & 127;   // reverse-roll(+4) == same mapping
          int ow = ((wc << 3) + (n & 7) + 4) & 255;
          size_t o = (((size_t)b * 32768) + oh * 256 + ow) * CDIM + cg;
          float gm = mods[b * 3072 + 1024 + cg];
          outf[o] = xres[o] + gm * (v + bias[cg]);
        } else {
          int t = rg, b = t >> 15;
          size_t o = (size_t)t * CDIM + cg;
          float gp = mods[b * 3072 + 2560 + cg];
          outf[o] = outf[o] + gp * v;
        }
      }
}

extern "C" void kernel_launch(void* const* d_in, const int* in_sizes, int n_in,
                              void* d_out, int out_size, void* d_ws, size_t ws_size,
                              hipStream_t stream) {
  const float* x           = (const float*)d_in[0];
  const float* c           = (const float*)d_in[1];
  const float* norm1_g     = (const float*)d_in[2];
  const float* norm1_b     = (const float*)d_in[3];
  const float* qkv_w       = (const float*)d_in[4];
  const float* q_bias      = (const float*)d_in[5];
  const float* v_bias      = (const float*)d_in[6];
  const float* logit_scale = (const float*)d_in[7];
  const float* cpb_w1      = (const float*)d_in[8];
  const float* cpb_b1      = (const float*)d_in[9];
  const float* cpb_w2      = (const float*)d_in[10];
  const float* proj_w      = (const float*)d_in[11];
  const float* proj_b      = (const float*)d_in[12];
  const float* norm2_g     = (const float*)d_in[13];
  const float* norm2_b     = (const float*)d_in[14];
  const float* fc1_w       = (const float*)d_in[15];
  const float* fc2_w       = (const float*)d_in[16];
  const float* ada_w       = (const float*)d_in[17];
  const float* ada_b       = (const float*)d_in[18];
  float* out = (float*)d_out;
  char*  ws  = (char*)d_ws;

  size_t off = 0;
  auto alloc = [&](size_t bytes) { size_t o = off; off += (bytes + 255) & ~(size_t)255; return o; };
  float* mods  = (float*)(ws + alloc((size_t)2 * 3072 * 4));
  float* biasq = (float*)(ws + alloc((size_t)1536 * 4));
  float* hbt   = (float*)(ws + alloc((size_t)225 * 16 * 4));
  float* rpb   = (float*)(ws + alloc((size_t)16 * 64 * 64 * 4));
  unsigned short* wq  = (unsigned short*)(ws + alloc((size_t)1536 * 512 * 2));
  unsigned short* wp  = (unsigned short*)(ws + alloc((size_t)512 * 512 * 2));
  unsigned short* wf1 = (unsigned short*)(ws + alloc((size_t)FC1N * 512 * 2));
  unsigned short* wf2 = (unsigned short*)(ws + alloc((size_t)512 * INNER_P * 2));
  unsigned short* r1  = (unsigned short*)(ws + alloc((size_t)TOK * 512 * 2));      // xw then h2
  unsigned short* r2  = (unsigned short*)(ws + alloc((size_t)TOK * 1536 * 2));     // qkv then hg
  unsigned short* r3  = (unsigned short*)(ws + alloc((size_t)TOK * 512 * 2));      // ctx
  unsigned short* r4  = (unsigned short*)(ws + alloc((size_t)TOK * FC1N * 2));     // fc1 out

  // prep
  mods_kernel<<<6144, 64, 0, stream>>>(c, ada_w, ada_b, mods);
  biasq_kernel<<<6, 256, 0, stream>>>(q_bias, v_bias, biasq);
  cvt_pad_kernel<<<(1536 * 512 + 255) / 256, 256, 0, stream>>>(qkv_w, wq, 1536, 512, 512);
  cvt_pad_kernel<<<(512 * 512 + 255) / 256, 256, 0, stream>>>(proj_w, wp, 512, 512, 512);
  cvt_pad_kernel<<<(FC1N * 512 + 255) / 256, 256, 0, stream>>>(fc1_w, wf1, FC1N, 512, 512);
  cvt_pad_kernel<<<(512 * INNER_P + 255) / 256, 256, 0, stream>>>(fc2_w, wf2, 512, INNER, INNER_P);
  cpb_kernel<<<225, 128, 0, stream>>>(cpb_w1, cpb_b1, cpb_w2, hbt);
  rpb_kernel<<<256, 256, 0, stream>>>(hbt, rpb);

  // attention branch
  ln_mod_kernel<<<TOK, 128, 0, stream>>>(x, norm1_g, norm1_b, mods, 512, 0, 1, r1);
  wmma_gemm_kernel<<<dim3(1536 / BN, TOK / BM), 256, 0, stream>>>(
      r1, 512, wq, 512, 1536, 512, 0, biasq, nullptr, nullptr, nullptr, r2, 1536);
  attn_kernel<<<16384, 64, 0, stream>>>(r2, rpb, logit_scale, r3);
  wmma_gemm_kernel<<<dim3(512 / BN, TOK / BM), 256, 0, stream>>>(
      r3, 512, wp, 512, 512, 512, 1, proj_b, x, mods, out, nullptr, 0);

  // GeGLU FFN branch
  ln_mod_kernel<<<TOK, 128, 0, stream>>>(out, norm2_g, norm2_b, mods, 2048, 1536, 0, r1);
  wmma_gemm_kernel<<<dim3((FC1N + BN - 1) / BN, TOK / BM), 256, 0, stream>>>(
      r1, 512, wf1, 512, FC1N, 512, 2, nullptr, nullptr, nullptr, nullptr, r4, FC1N);
  geglu_kernel<<<(int)(((size_t)TOK * INNER_P) / 256), 256, 0, stream>>>(r4, r2);
  wmma_gemm_kernel<<<dim3(512 / BN, TOK / BM), 256, 0, stream>>>(
      r2, INNER_P, wf2, INNER_P, 512, INNER_P, 3, nullptr, nullptr, mods, out, nullptr, 0);
}
